// BinaryBlock_309237646124
// MI455X (gfx1250) — 8-bit matrix operand order repaired and checked by reading, compiled, not run on hardware
//
#include <hip/hip_runtime.h>
#include <cstdint>

#define BB 16
#define CC 128
#define HH 128
#define WW 128
#define CO 64
#define KTOT (CC * 9)

typedef __attribute__((ext_vector_type(8))) int v8i;

#define ACT_BYTES  (size_t)(BB * HH * WW * CC)
#define SC_BYTES   (size_t)(BB * HH * WW * CO * 4)
#define WP_BYTES   (size_t)(2 * 9 * 64 * 64)

__global__ __launch_bounds__(128)
void prep_act_kernel(const float* __restrict__ x,
                     const float* __restrict__ bias1,
                     const float* __restrict__ prelu_a,
                     const float* __restrict__ bias2,
                     const float* __restrict__ pool_w,
                     int8_t* __restrict__ act,
                     float*  __restrict__ sc) {
    const int bh = blockIdx.x;
    const int b  = bh >> 7;
    const int h  = bh & (HH - 1);
    const int w  = threadIdx.x;
    const size_t pix = (size_t)bh * WW + w;

    int8_t sv[CC];
    float  scv[CO];
    float  xprev = 0.0f;

    #pragma unroll
    for (int c = 0; c < CC; ++c) {
        const float xv = x[(((size_t)b * CC + c) * HH + h) * WW + w];
        float t = xv + bias1[c];
        t = fmaxf(t, 0.0f) + prelu_a[c] * fminf(t, 0.0f);
        t = t + bias2[c];
        sv[c] = (t > 0.0f) ? (int8_t)1 : ((t < 0.0f) ? (int8_t)-1 : (int8_t)0);
        if (c & 1) {
            const int o = c >> 1;
            scv[o] = pool_w[2 * o] * xprev + pool_w[2 * o + 1] * xv;
        } else {
            xprev = xv;
        }
    }

    int4* ad = (int4*)(act + pix * CC);
    const int4* as = (const int4*)sv;
    #pragma unroll
    for (int i = 0; i < 8; ++i) ad[i] = as[i];

    float4* sd = (float4*)(sc + pix * CO);
    const float4* ss = (const float4*)scv;
    #pragma unroll
    for (int i = 0; i < 16; ++i) sd[i] = ss[i];
}

__global__ __launch_bounds__(128)
void prep_w_kernel(const float* __restrict__ cw,
                   int8_t* __restrict__ wp,
                   float*  __restrict__ sf) {
    __shared__ float red[128];
    const int o = blockIdx.x;
    const int t = threadIdx.x;

    float s = 0.0f;
    for (int j = t; j < KTOT; j += 128) {
        const float v = cw[(size_t)o * KTOT + j];
        s += fabsf(v);
        const int c   = j / 9;
        const int tap = j - 9 * c;
        const int ch  = c >> 6;
        const int cc  = c & 63;
        wp[(((size_t)ch * 9 + tap) * 64 + o) * 64 + cc] =
            (v > 0.0f) ? (int8_t)1 : ((v < 0.0f) ? (int8_t)-1 : (int8_t)0);
    }
    red[t] = s;
    __syncthreads();
    #pragma unroll
    for (int k = 64; k > 0; k >>= 1) {
        if (t < k) red[t] += red[t + k];
        __syncthreads();
    }
    if (t == 0) sf[o] = red[0] * (1.0f / (float)KTOT);
}

__global__ __launch_bounds__(256)
void conv_wmma_kernel(const int8_t* __restrict__ act,
                      const int8_t* __restrict__ wp,
                      const float*  __restrict__ sf,
                      const float*  __restrict__ sc,
                      float* __restrict__ out) {
    __shared__ __align__(16) char smem[9 * 64 * 64];
    int8_t* lw    = (int8_t*)smem;
    float*  ytile = (float*)smem;

    const int tid  = threadIdx.x;
    const int wave = tid >> 5;
    const int lane = tid & 31;
    const int half = lane >> 4;
    const int l16  = lane & 15;
    const int koff = half * 16;

    const int bh = blockIdx.x;
    const int b  = bh >> 7;
    const int h  = bh & (HH - 1);
    const int w0 = wave * 16;
    const int p  = w0 + l16;

    v8i acc[4];
    #pragma unroll
    for (int nt = 0; nt < 4; ++nt) acc[nt] = (v8i)(0);

    union AFrag { uint64_t q[4]; v8i v; };
    union BFrag { int4 x[2];    v8i v; };

    for (int ch = 0; ch < 2; ++ch) {
        __syncthreads();
        {
            const int4* src = (const int4*)(wp + (size_t)ch * (9 * 64 * 64));
            int4* dst = (int4*)lw;
            for (int i = tid; i < (9 * 64 * 64) / 16; i += 256) dst[i] = src[i];
        }
        __syncthreads();

        #pragma unroll
        for (int kh = 0; kh < 3; ++kh) {
            const int hi = h + kh - 1;
            if (hi < 0 || hi >= HH) continue;
            const size_t rowbase = ((size_t)(b * HH + hi) * WW) * CC;
            #pragma unroll
            for (int kw = 0; kw < 3; ++kw) {
                const int  wi  = p + kw - 1;
                const bool inb = (wi >= 0) && (wi < WW);
                const int8_t* ap =
                    act + rowbase + (size_t)(inb ? wi : 0) * CC + ch * 64 + koff;

                AFrag a;
                const uint64_t* qp = (const uint64_t*)ap;
                uint64_t d0 = qp[0], d1 = qp[1], d2 = qp[4], d3 = qp[5];
                if (!inb) { d0 = 0; d1 = 0; d2 = 0; d3 = 0; }
                a.q[0] = d0; a.q[1] = d1; a.q[2] = d2; a.q[3] = d3;

                const int tap = kh * 3 + kw;
                #pragma unroll
                for (int nt = 0; nt < 4; ++nt) {
                    const int o = l16 + 16 * nt;
                    const int8_t* bp = lw + ((size_t)tap * 64 + o) * 64 + half * 16;
                    BFrag bf;
                    bf.x[0] = *(const int4*)(bp);
                    bf.x[1] = *(const int4*)(bp + 32);
                    acc[nt] = __builtin_amdgcn_wmma_i32_16x16x64_iu8(
                         true, a.v,  true, bf.v,
                        acc[nt],  false,  false);
                }
            }
        }
    }

    __syncthreads();
    const size_t pixrow = (size_t)(b * HH + h) * WW;
    #pragma unroll
    for (int nt = 0; nt < 4; ++nt) {
        const int o = l16 + 16 * nt;
        const float s = sf[o];
        #pragma unroll
        for (int r = 0; r < 8; ++r) {
            const int m = r + half * 8;
            const int w = w0 + m;
            ytile[w * 65 + o] = s * (float)acc[nt][r] + sc[(pixrow + w) * CO + o];
        }
    }
    __syncthreads();

    {
        const int seg = tid >> 1;
        const int oo  = seg >> 1;
        const int s2  = seg & 1;
        const int sub = tid & 1;
        const int cout = oo * 4 + (h & 1) * 2 + s2;
        float* obase =
            out + (((size_t)b * (CO * 4) + cout) * (HH / 2) + (h >> 1)) * (WW / 2);
        #pragma unroll
        for (int j = 0; j < 8; ++j) {
            const int wo = sub * 32 + j * 4;
            float4 v;
            v.x = ytile[(2 * (wo + 0) + s2) * 65 + oo];
            v.y = ytile[(2 * (wo + 1) + s2) * 65 + oo];
            v.z = ytile[(2 * (wo + 2) + s2) * 65 + oo];
            v.w = ytile[(2 * (wo + 3) + s2) * 65 + oo];
            *(float4*)(obase + wo) = v;
        }
    }
}

extern "C" void kernel_launch(void* const* d_in, const int* in_sizes, int n_in,
                              void* d_out, int out_size, void* d_ws, size_t ws_size,
                              hipStream_t stream) {
    const float* x       = (const float*)d_in[0];
    const float* bias1   = (const float*)d_in[1];
    const float* prelu_a = (const float*)d_in[2];
    const float* bias2   = (const float*)d_in[3];
    const float* conv_w  = (const float*)d_in[4];
    const float* pool_w  = (const float*)d_in[5];

    char* ws = (char*)d_ws;
    int8_t* act = (int8_t*)ws;
    float*  sc  = (float*)(ws + ACT_BYTES);
    int8_t* wp  = (int8_t*)(ws + ACT_BYTES + SC_BYTES);
    float*  sf  = (float*)(ws + ACT_BYTES + SC_BYTES + WP_BYTES);

    prep_act_kernel<<<BB * HH, 128, 0, stream>>>(x, bias1, prelu_a, bias2, pool_w,
                                                 act, sc);
    prep_w_kernel<<<CO, 128, 0, stream>>>(conv_w, wp, sf);
    conv_wmma_kernel<<<BB * HH, 256, 0, stream>>>(act, wp, sf, sc, (float*)d_out);
}
